// HybridUT_19404662243465
// MI455X (gfx1250) — matrix operand order repaired and checked by reading, not compiled, not run on hardware
//
#include <hip/hip_runtime.h>

// ---------------------------------------------------------------------------
// HybridUT attention for MI455X (gfx1250, wave32, WMMA).
// f16 intermediates laid out so every WMMA fragment is 2x16B contiguous loads;
// async global->LDS staging (guarded) for the attention K/V tiles.
// ---------------------------------------------------------------------------

typedef __attribute__((ext_vector_type(16))) _Float16 v16h;
typedef __attribute__((ext_vector_type(8)))  _Float16 v8h;
typedef __attribute__((ext_vector_type(8)))  float    v8f;

#define NB    8
#define NC    256
#define NHW   4096
#define HEADS 8
#define DHEAD 32
#define PJN   256      // RS*RS
#define KLD   40       // padded ld (halves) for kn  [j][d]
#define VLD   264      // padded ld (halves) for v^T [d][j]

#if defined(__has_builtin)
#if __has_builtin(__builtin_amdgcn_global_load_async_to_lds_b128) && \
    __has_builtin(__builtin_amdgcn_s_wait_asynccnt)
#define USE_ASYNC_LDS 1
#else
#define USE_ASYNC_LDS 0
#endif
#else
#define USE_ASYNC_LDS 0
#endif

#if USE_ASYNC_LDS
typedef int nvi4 __attribute__((vector_size(16)));       // native clang int vector
typedef __attribute__((address_space(1))) nvi4* gptr_i4; // global 128b vector ptr
typedef __attribute__((address_space(3))) nvi4* lptr_i4; // LDS    128b vector ptr
#endif

static __device__ inline void cp16B(const _Float16* g, _Float16* l)
{
#if USE_ASYNC_LDS
    __builtin_amdgcn_global_load_async_to_lds_b128((gptr_i4)g, (lptr_i4)l, 0, 0);
#else
    *(v8h*)l = *(const v8h*)g;
#endif
}
static __device__ inline void cp_wait()
{
#if USE_ASYNC_LDS
    __builtin_amdgcn_s_wait_asynccnt(0);
#endif
    __syncthreads();
}

// ---- fragment helpers (CDNA5 16-bit WMMA layouts, contiguous 16B runs) ----
static __device__ inline v16h join8(v8h lo, v8h hi8)
{
    return __builtin_shufflevector(lo, hi8, 0, 1, 2, 3, 4, 5, 6, 7,
                                   8, 9, 10, 11, 12, 13, 14, 15);
}
// A 16x32 f16: lane(row r, half hi) reads k runs [hi*8, hi*8+8) and +16.
static __device__ inline v16h frag_a16(const _Float16* row, int hi)
{
    v8h lo = *(const v8h*)(row + hi * 8);
    v8h up = *(const v8h*)(row + 16 + hi * 8);
    return join8(lo, up);
}
// B 32x16 f16 from "n-major" storage: nrow -> contiguous k; lane reads k runs [hi*8, hi*8+8) and +16.
static __device__ inline v16h frag_b16(const _Float16* nrow, int hi)
{
    v8h lo = *(const v8h*)(nrow + hi * 8);
    v8h up = *(const v8h*)(nrow + 16 + hi * 8);
    return join8(lo, up);
}
// A fragment from f32 source (attn rows), converted on the fly.
static __device__ inline v16h frag_a32(const float* row, int hi)
{
    const float4 f0 = *(const float4*)(row + hi * 8);
    const float4 f1 = *(const float4*)(row + hi * 8 + 4);
    const float4 f2 = *(const float4*)(row + 16 + hi * 8);
    const float4 f3 = *(const float4*)(row + 16 + hi * 8 + 4);
    v16h a;
    a[0] = (_Float16)f0.x;  a[1] = (_Float16)f0.y;
    a[2] = (_Float16)f0.z;  a[3] = (_Float16)f0.w;
    a[4] = (_Float16)f1.x;  a[5] = (_Float16)f1.y;
    a[6] = (_Float16)f1.z;  a[7] = (_Float16)f1.w;
    a[8] = (_Float16)f2.x;  a[9] = (_Float16)f2.y;
    a[10] = (_Float16)f2.z; a[11] = (_Float16)f2.w;
    a[12] = (_Float16)f3.x; a[13] = (_Float16)f3.y;
    a[14] = (_Float16)f3.z; a[15] = (_Float16)f3.w;
    return a;
}

// ---------------- weight f32 -> f16 ----------------------------------------
__global__ void __launch_bounds__(256) cvt_f32_f16(const float* __restrict__ s,
                                                   _Float16* __restrict__ d)
{
    int i = blockIdx.x * 256 + threadIdx.x;
    d[i] = (_Float16)s[i];
}

// ---------------- depthwise 3x3 conv, pad 1; f32 in, f16 out ---------------
__global__ void __launch_bounds__(256) dwconv3x3(const float* __restrict__ x,
                                                 const float* __restrict__ w,
                                                 _Float16* __restrict__ y)
{
    int idx = blockIdx.x * 256 + threadIdx.x;         // exact NB*NC*NHW/256
    int px = idx & 63, py = (idx >> 6) & 63;
    int c  = (idx >> 12) & 255, b = idx >> 20;
    const float* xp = x + ((size_t)(b * NC + c)) * NHW;
    const float* wp = w + c * 9;
    float acc = 0.f;
#pragma unroll
    for (int ky = 0; ky < 3; ++ky) {
        int yy = py + ky - 1;
        if ((unsigned)yy < 64u) {
#pragma unroll
            for (int kx = 0; kx < 3; ++kx) {
                int xx = px + kx - 1;
                if ((unsigned)xx < 64u) acc += xp[yy * 64 + xx] * wp[ky * 3 + kx];
            }
        }
    }
    y[idx] = (_Float16)acc;
}

// ---------------- pointwise conv as GEMM: Y = W(f16) * X(f16) --------------
// 8 waves; block tile 32(M) x 256(N); wave tile 16x64 (4 WMMA / K-step).
// A frags straight from global; B staged transposed in LDS (n-major, ld=40).
__global__ void __launch_bounds__(256) pw_gemm_wmma(const _Float16* __restrict__ Wgt,
                                                    const _Float16* __restrict__ X,
                                                    float* __restrict__ Y,
                                                    int M, int K, int N,
                                                    long xStride, long yStride)
{
    __shared__ _Float16 sBt[256 * KLD];
    int tid = threadIdx.x, lane = tid & 31, wave = tid >> 5;
    int hi = lane >> 4, r = lane & 15, n = lane & 15;
    int wm = wave >> 2, wn = wave & 3;
    int n0 = blockIdx.x * 256;
    int m0 = blockIdx.y * 32;
    const _Float16* Xb = X + (size_t)blockIdx.z * xStride;
    float*          Yb = Y + (size_t)blockIdx.z * yStride;
    const _Float16* arow = Wgt + (size_t)(m0 + wm * 16 + r) * K;

    v8f acc[4] = {};
    for (int k0 = 0; k0 < K; k0 += 32) {
#pragma unroll
        for (int u = 0; u < 4; ++u) {                  // stage B^T: 32x256 tile
            int e = u * 256 + tid;                     // 1024 v8h units
            int k = e >> 5, n8 = (e & 31) * 8;
            v8h vv = *(const v8h*)&Xb[(size_t)(k0 + k) * N + n0 + n8];
#pragma unroll
            for (int q = 0; q < 8; ++q) sBt[(n8 + q) * KLD + k] = vv[q];
        }
        if (k0 + 32 < K)
            __builtin_prefetch(&Xb[(size_t)(k0 + 32) * N + n0 + tid * 8], 0, 0);
        __syncthreads();
        v16h a = frag_a16(arow + k0, hi);
        const _Float16* bbase = &sBt[(wn * 64 + n) * KLD];
#pragma unroll
        for (int t = 0; t < 4; ++t) {
            v16h b = frag_b16(bbase + t * 16 * KLD, hi);
            acc[t] = __builtin_amdgcn_wmma_f32_16x16x32_f16(false, a, false, b,
                                                            (short)0, acc[t],
                                                            false, false);
        }
        __syncthreads();
    }
#pragma unroll
    for (int t = 0; t < 4; ++t)
#pragma unroll
        for (int i = 0; i < 8; ++i) {
            int m = hi * 8 + i;
            Yb[(size_t)(m0 + wm * 16 + m) * N + n0 + wn * 64 + t * 16 + n] = acc[t][i];
        }
}

// ---------------- bilinear 64x64 -> 16x16, align_corners=True --------------
__global__ void __launch_bounds__(256) bilinear_ds(const float* __restrict__ qkv,
                                                   int choff, float* __restrict__ out)
{
    int idx = blockIdx.x * 256 + threadIdx.x;          // NB*NC*PJN exact
    int pj = idx & 255, c = (idx >> 8) & 255, b = idx >> 16;
    int oy = pj >> 4, ox = pj & 15;
    const float scale = 63.0f / 15.0f;
    float fy = oy * scale, fx = ox * scale;
    int y0 = (int)floorf(fy), x0 = (int)floorf(fx);
    int y1 = min(y0 + 1, 63), x1 = min(x0 + 1, 63);
    float wy = fy - (float)y0, wx = fx - (float)x0;
    const float* s = qkv + ((size_t)(b * 768 + choff + c)) * NHW;
    float top = s[y0 * 64 + x0] * (1.f - wx) + s[y0 * 64 + x1] * wx;
    float bot = s[y1 * 64 + x0] * (1.f - wx) + s[y1 * 64 + x1] * wx;
    out[idx] = top * (1.f - wy) + bot * wy;
}

// ---------------- q: head split + LayerNorm(dh=32) -> f16 ------------------
__global__ void __launch_bounds__(256) head_ln_q(const float* __restrict__ qkv,
                                                 const float* __restrict__ g,
                                                 const float* __restrict__ be,
                                                 _Float16* __restrict__ qn)
{
    int p = blockIdx.x & 4095, b = blockIdx.x >> 12;
    int h = threadIdx.x >> 5, d = threadIdx.x & 31;
    float v = qkv[((size_t)(b * 768) + (d * HEADS + h)) * NHW + p];
    float s = v;
#pragma unroll
    for (int o = 16; o > 0; o >>= 1) s += __shfl_xor(s, o, 32);
    float dv = v - s * (1.f / 32.f);
    float q2 = dv * dv;
#pragma unroll
    for (int o = 16; o > 0; o >>= 1) q2 += __shfl_xor(q2, o, 32);
    float inv = rsqrtf(q2 * (1.f / 32.f) + 1e-6f);
    qn[((size_t)(b * HEADS + h) * NHW + p) * DHEAD + d] =
        (_Float16)(dv * inv * g[h * DHEAD + d] + be[h * DHEAD + d]);
}

// ---------------- k/v head split; k gets LN -> kn[j][d] ld40 (f16),
// ---------------- v -> v^T[d][j] ld264 (f16) -------------------------------
__global__ void __launch_bounds__(256) head_kv(const float* __restrict__ src,
                                               const float* __restrict__ g,
                                               const float* __restrict__ be,
                                               _Float16* __restrict__ dst, int mode)
{
    int pj = blockIdx.x & 255, b = blockIdx.x >> 8;
    int h = threadIdx.x >> 5, d = threadIdx.x & 31;
    float v = src[((size_t)(b * NC) + (d * HEADS + h)) * PJN + pj];
    if (mode == 0) {
        float s = v;
#pragma unroll
        for (int o = 16; o > 0; o >>= 1) s += __shfl_xor(s, o, 32);
        float dv = v - s * (1.f / 32.f);
        float q2 = dv * dv;
#pragma unroll
        for (int o = 16; o > 0; o >>= 1) q2 += __shfl_xor(q2, o, 32);
        float inv = rsqrtf(q2 * (1.f / 32.f) + 1e-6f);
        v = dv * inv * g[h * DHEAD + d] + be[h * DHEAD + d];
        dst[((size_t)(b * HEADS + h) * PJN + pj) * KLD + d] = (_Float16)v;
    } else {
        dst[((size_t)(b * HEADS + h) * DHEAD + d) * VLD + pj] = (_Float16)v;
    }
}

// ---------------- attn = (qn . kn^T + bias)/32 -----------------------------
__global__ void __launch_bounds__(256) attn_wmma(const _Float16* __restrict__ qn,
                                                 const _Float16* __restrict__ kn,
                                                 const float* __restrict__ table,
                                                 const int* __restrict__ rel,
                                                 float* __restrict__ attn)
{
    __shared__ _Float16 sK[PJN * KLD];                 // linear mirror of kn[bh]
    int tid = threadIdx.x, lane = tid & 31, wave = tid >> 5;
    int bh = blockIdx.y, h = bh & 7;
    const _Float16* knb = kn + (size_t)bh * PJN * KLD;
#pragma unroll
    for (int u = 0; u < 5; ++u) {                      // 1280 x 16B async copies
        int o = (u * 256 + tid) * 8;
        cp16B(knb + o, &sK[o]);
    }
    cp_wait();

    int i0 = (blockIdx.x * 8 + wave) * 16;
    const _Float16* qrow = qn + ((size_t)bh * NHW + i0) * DHEAD;
    int r = lane & 15, hi = lane >> 4, n = lane & 15;
    v16h a = frag_a16(qrow + r * DHEAD, hi);
    for (int jt = 0; jt < 16; ++jt) {
        v16h bb = frag_b16(&sK[(jt * 16 + n) * KLD], hi);
        v8f c = {};
        c = __builtin_amdgcn_wmma_f32_16x16x32_f16(false, a, false, bb,
                                                   (short)0, c, false, false);
#pragma unroll
        for (int i = 0; i < 8; ++i) {
            int m = hi * 8 + i;
            int p = i0 + m, j = jt * 16 + n;
            int py = p >> 6, px = p & 63;
            int cell = ((py >> 2) << 4) + (px >> 2);
            int ridx = rel[cell * PJN + j];
            float bias = table[ridx * HEADS + h];
            attn[((size_t)bh * NHW + p) * PJN + j] = (c[i] + bias) * (1.f / 32.f);
        }
    }
}

// ---------------- out = attn . v  -> NCHW ----------------------------------
__global__ void __launch_bounds__(256) attnv_wmma(const float* __restrict__ attn,
                                                  const _Float16* __restrict__ vhT,
                                                  float* __restrict__ o)
{
    __shared__ _Float16 sV[DHEAD * VLD];               // linear mirror of v^T[bh]
    int tid = threadIdx.x, lane = tid & 31, wave = tid >> 5;
    int bh = blockIdx.y, b = bh >> 3, h = bh & 7;
    const _Float16* vb = vhT + (size_t)bh * DHEAD * VLD;
    for (int u = tid; u < (DHEAD * VLD) / 8; u += 256) // 1056 x 16B copies
        cp16B(vb + u * 8, &sV[u * 8]);
    cp_wait();

    int i0 = (blockIdx.x * 8 + wave) * 16;
    const float* arow = attn + ((size_t)bh * NHW + i0) * PJN;
    int r = lane & 15, hi = lane >> 4, n = lane & 15;
    v8f c0 = {}, c1 = {};
    for (int k0 = 0; k0 < PJN; k0 += 32) {
        v16h a  = frag_a32(arow + r * PJN + k0, hi);
        v16h b0 = frag_b16(&sV[n * VLD + k0], hi);
        v16h b1 = frag_b16(&sV[(16 + n) * VLD + k0], hi);
        c0 = __builtin_amdgcn_wmma_f32_16x16x32_f16(false, a, false, b0,
                                                    (short)0, c0, false, false);
        c1 = __builtin_amdgcn_wmma_f32_16x16x32_f16(false, a, false, b1,
                                                    (short)0, c1, false, false);
    }
#pragma unroll
    for (int i = 0; i < 8; ++i) {
        int m = hi * 8 + i, p = i0 + m;
        o[((size_t)(b * NC) + (n * HEADS + h)) * NHW + p] = c0[i];
        o[((size_t)(b * NC) + ((16 + n) * HEADS + h)) * NHW + p] = c1[i];
    }
}

// ---------------------------------------------------------------------------
extern "C" void kernel_launch(void* const* d_in, const int* in_sizes, int n_in,
                              void* d_out, int out_size, void* d_ws, size_t ws_size,
                              hipStream_t stream)
{
    (void)in_sizes; (void)n_in; (void)out_size; (void)ws_size;
    const float* x       = (const float*)d_in[0];
    const float* qkv_dw  = (const float*)d_in[1];
    const float* qkv_pw  = (const float*)d_in[2];
    const float* out_dw  = (const float*)d_in[3];
    const float* out_pw  = (const float*)d_in[4];
    const float* gq      = (const float*)d_in[5];
    const float* bq      = (const float*)d_in[6];
    const float* gk      = (const float*)d_in[7];
    const float* bk      = (const float*)d_in[8];
    const float* table   = (const float*)d_in[9];
    const int*   rel     = (const int*)d_in[10];

    float* out0 = (float*)d_out;                       // [8,256,64,64]
    float* attn = (float*)d_out + 8388608;             // [8,8,4096,256]

    char* wsb = (char*)d_ws;                           // ~135 MB peak (aliased)
    float*    qkv  = (float*)(wsb + 0);                // 100.7 MB
    float*    o    = (float*)(wsb + 0);                // alias (qkv dead by then)
    _Float16* t1   = (_Float16*)(wsb + 100663296);     // 16.8 MB, reused as t2
    _Float16* qn   = (_Float16*)(wsb + 117440512);     // 16.8 MB
    float*    kd   = (float*)(wsb + 134217728);        // 2 MB
    float*    vd   = (float*)(wsb + 136314880);        // 2 MB
    _Float16* kn   = (_Float16*)(wsb + 138412032);     // 1.3 MB (ld=40)
    _Float16* vhT  = (_Float16*)(wsb + 139722752);     // 1.1 MB (ld=264)
    _Float16* wq16 = (_Float16*)(wsb + 140804096);     // 0.4 MB
    _Float16* wo16 = (_Float16*)(wsb + 141197312);     // 0.13 MB

    // 0) weights to f16
    cvt_f32_f16<<<768, 256, 0, stream>>>(qkv_pw, wq16);
    cvt_f32_f16<<<256, 256, 0, stream>>>(out_pw, wo16);
    // 1) qkv = pw(dw(x))
    dwconv3x3<<<32768, 256, 0, stream>>>(x, qkv_dw, t1);
    pw_gemm_wmma<<<dim3(16, 24, NB), 256, 0, stream>>>(wq16, t1, qkv,
        768, 256, 4096, 256L * 4096, 768L * 4096);
    // 2) downsample k, v
    bilinear_ds<<<2048, 256, 0, stream>>>(qkv, 256, kd);
    bilinear_ds<<<2048, 256, 0, stream>>>(qkv, 512, vd);
    // 3) head split + LayerNorm -> f16 WMMA-friendly layouts
    head_ln_q<<<32768, 256, 0, stream>>>(qkv, gq, bq, qn);
    head_kv<<<2048, 256, 0, stream>>>(kd, gk, bk, kn, 0);
    head_kv<<<2048, 256, 0, stream>>>(vd, gk, bk, vhT, 1);
    // 4) attn -> second output region
    attn_wmma<<<dim3(32, 64), 256, 0, stream>>>(qn, kn, table, rel, attn);
    // 5) out = attn.v -> NCHW (reuses qkv region)
    attnv_wmma<<<dim3(32, 64), 256, 0, stream>>>(attn, vhT, o);
    // 6) final pw(dw(o)) -> first output region
    dwconv3x3<<<32768, 256, 0, stream>>>(o, out_dw, t1);
    pw_gemm_wmma<<<dim3(16, 8, NB), 256, 0, stream>>>(wo16, t1, out0,
        256, 256, 4096, 256L * 4096, 256L * 4096);
}
